// ResNetBlock_59854664237618
// MI455X (gfx1250) — compile-verified, matrix operand order checked by reading, not run on hardware
//
#include <hip/hip_runtime.h>

// Problem constants (from the reference)
#define VCOUNT 50000
#define RDIM   5
#define ADIM   8
#define CDIM   32
#define TDIM   32
#define NROT   4
#define DELTA  2

typedef __attribute__((ext_vector_type(16))) __bf16 bf16x16;
typedef __attribute__((ext_vector_type(8)))  __bf16 bf16x8;
typedef __attribute__((ext_vector_type(4)))  __bf16 bf16x4;
typedef __attribute__((ext_vector_type(8)))  float  f32x8;
typedef __attribute__((ext_vector_type(4)))  float  f32x4;

// float -> bf16, round-to-nearest-even (pure integer ops; always compiles)
__device__ __forceinline__ __bf16 f2bf(float f) {
    unsigned u = __builtin_bit_cast(unsigned, f);
    unsigned r = u + 0x7FFFu + ((u >> 16) & 1u);
    unsigned short h = (unsigned short)(r >> 16);
    return __builtin_bit_cast(__bf16, h);
}

__device__ __forceinline__ f32x8 wmma_bf16(bf16x16 a, bf16x16 b, f32x8 c) {
    // (neg_a, A, neg_b, B, c_mod, C, reuse_a, reuse_b)
    return __builtin_amdgcn_wmma_f32_16x16x32_bf16(false, a, false, b, (short)0, c,
                                                   false, false);
}

// One Dirac-conv pass: gather+barycentric interp, WMMA contraction against the
// template bank for all 4 rotations, angular max.
//   mode 0: out = max_n relu(conv + bias)              (conv1)
//   mode 1: out = relu(max_n(conv + bias) + residual)  (conv2 + residual + relu)
// Per 256-thread block: 8 waves x 16 vertices = 128 vertices.
__global__ void __launch_bounds__(256, 1)
dirac_conv_kernel(const float* __restrict__ x,
                  const int*   __restrict__ bc_idx,
                  const float* __restrict__ bc_w,
                  const float* __restrict__ templ,
                  const float* __restrict__ bias,
                  const float* __restrict__ residual,
                  float*       __restrict__ out,
                  int mode)
{
    extern __shared__ __align__(16) char smem[];
    __bf16* sB = (__bf16*)smem;                                   // [R][A][T][C] bf16 = 80KB
    __bf16* sA = (__bf16*)(smem + RDIM * ADIM * TDIM * CDIM * 2); // 8 waves x [A][16][C] bf16

    const int tid  = threadIdx.x;
    const int lane = tid & 31;
    const int wv   = tid >> 5;
    const int hlf  = lane >> 4;  // 0: K 0..7 / 16..23  1: K 8..15 / 24..31
    const int mrow = lane & 15;

    // ---- Stage templates into LDS as bf16, transposed [t][r][a][c] -> [r][a][t][c] ----
    // (B-matrix K-blocks are (r,a); within a block K = channel, N = template.)
    #pragma unroll 4
    for (int o4 = tid; o4 < (RDIM * ADIM * TDIM * CDIM) / 4; o4 += 256) {
        int c4 = o4 & 7;
        int t  = (o4 >> 3) & 31;
        int a  = (o4 >> 8) & 7;
        int r  = o4 >> 11;
        f32x4 v = *(const f32x4*)(templ + (((t * RDIM + r) * ADIM + a) * CDIM + c4 * 4));
        bf16x4 h;
        h.x = f2bf(v.x); h.y = f2bf(v.y); h.z = f2bf(v.z); h.w = f2bf(v.w);
        *(bf16x4*)(sB + o4 * 4) = h;
    }
    __syncthreads();

    int  vbase  = (blockIdx.x * 8 + wv) * 16;
    bool active = (vbase < VCOUNT);          // V % 16 == 0 -> whole tile valid or invalid
    if (!active) vbase = 0;

    __bf16* sAw = sA + wv * (ADIM * 16 * CDIM);

    f32x8 acc[NROT][2];
    #pragma unroll
    for (int n = 0; n < NROT; ++n)
        #pragma unroll
        for (int q = 0; q < 2; ++q)
            #pragma unroll
            for (int e = 0; e < 8; ++e) acc[n][q][e] = 0.0f;

    for (int r = 0; r < RDIM; ++r) {
        // ---- Phase 1: barycentric interpolation for one radial ring (lane = channel) ----
        for (int a = 0; a < ADIM; ++a) {
            if (a + 1 < ADIM) {   // prefetch next angular step's tables (global_prefetch_b8)
                long pb = (((long)vbase * RDIM + r) * ADIM + (a + 1)) * 3;
                __builtin_prefetch(bc_idx + pb, 0, 3);
                __builtin_prefetch(bc_w + pb, 0, 3);
            }
            #pragma unroll 4
            for (int m = 0; m < 16; ++m) {
                long b  = (((long)(vbase + m) * RDIM + r) * ADIM + a) * 3;
                int   i0 = bc_idx[b + 0], i1 = bc_idx[b + 1], i2 = bc_idx[b + 2];
                float w0 = bc_w[b + 0],   w1 = bc_w[b + 1],   w2 = bc_w[b + 2];
                float val = w0 * x[i0 * CDIM + lane]
                          + w1 * x[i1 * CDIM + lane]
                          + w2 * x[i2 * CDIM + lane];
                sAw[(a * 16 + m) * CDIM + lane] = f2bf(val);
            }
        }
        __syncthreads();

        // ---- Phase 2: load the 8 A fragments (ISA 16x32 bf16 A layout) ----
        bf16x16 af[ADIM];
        #pragma unroll
        for (int a = 0; a < ADIM; ++a) {
            const __bf16* p = sAw + (a * 16 + mrow) * CDIM + hlf * 8;
            ((bf16x8*)&af[a])[0] = *(const bf16x8*)(p);       // K = hlf*8 .. +7
            ((bf16x8*)&af[a])[1] = *(const bf16x8*)(p + 16);  // K = 16+hlf*8 .. +7
        }

        // ---- 8 template K-blocks x 4 rotations x 2 N-tiles of WMMA ----
        // roll(templ,+n*DELTA) correlation == pairing templ block j with interp
        // block (j + n*DELTA) mod A, so all rotations share the A fragments.
        #pragma unroll
        for (int j = 0; j < ADIM; ++j) {
            const __bf16* q = sB + ((r * ADIM + j) * TDIM + mrow) * CDIM + hlf * 8;
            bf16x16 b0, b1;
            ((bf16x8*)&b0)[0] = *(const bf16x8*)(q);
            ((bf16x8*)&b0)[1] = *(const bf16x8*)(q + 16);
            ((bf16x8*)&b1)[0] = *(const bf16x8*)(q + 16 * CDIM);
            ((bf16x8*)&b1)[1] = *(const bf16x8*)(q + 16 * CDIM + 16);
            #pragma unroll
            for (int n = 0; n < NROT; ++n) {
                int a = (j + DELTA * n) & (ADIM - 1);
                acc[n][0] = wmma_bf16(af[a], b0, acc[n][0]);
                acc[n][1] = wmma_bf16(af[a], b1, acc[n][1]);
            }
        }
        __syncthreads();
    }

    // ---- Epilogue: bias, ReLU/angular-max (max∘relu == relu∘max), residual ----
    float bv[2] = { bias[mrow], bias[16 + mrow] };
    #pragma unroll
    for (int nt = 0; nt < 2; ++nt) {
        #pragma unroll
        for (int e = 0; e < 8; ++e) {
            float mval = acc[0][nt][e] + bv[nt];
            #pragma unroll
            for (int n = 1; n < NROT; ++n) mval = fmaxf(mval, acc[n][nt][e] + bv[nt]);
            int m   = hlf * 8 + e;                 // C/D layout: VGPR e -> row m
            int idx = (vbase + m) * TDIM + nt * 16 + mrow;
            if (mode == 0) {
                mval = fmaxf(mval, 0.0f);
                if (active) out[idx] = mval;
            } else {
                mval = fmaxf(mval + residual[idx], 0.0f);
                if (active) out[idx] = mval;
            }
        }
    }
}

extern "C" void kernel_launch(void* const* d_in, const int* in_sizes, int n_in,
                              void* d_out, int out_size, void* d_ws, size_t ws_size,
                              hipStream_t stream) {
    (void)in_sizes; (void)n_in; (void)out_size; (void)ws_size;
    const float* signal = (const float*)d_in[0];
    const int*   bc_idx = (const int*)d_in[1];
    const float* bc_w   = (const float*)d_in[2];
    const float* templ1 = (const float*)d_in[3];
    const float* bias1  = (const float*)d_in[4];
    const float* templ2 = (const float*)d_in[5];
    const float* bias2  = (const float*)d_in[6];
    float* out = (float*)d_out;
    float* s1  = (float*)d_ws;   // intermediate [V, T] f32 (6.4 MB)

    const int smem = RDIM * ADIM * TDIM * CDIM * 2   // template bank, bf16
                   + 8 * ADIM * 16 * CDIM * 2;       // per-wave interp tiles, bf16
    dim3 grid((VCOUNT + 127) / 128), block(256);

    dirac_conv_kernel<<<grid, block, smem, stream>>>(
        signal, bc_idx, bc_w, templ1, bias1, nullptr, s1, 0);
    dirac_conv_kernel<<<grid, block, smem, stream>>>(
        s1, bc_idx, bc_w, templ2, bias2, signal, out, 1);
}
